// MultiHeadAttention_10883447128368
// MI455X (gfx1250) — compile-verified, matrix operand order checked by reading, not run on hardware
//
#include <hip/hip_runtime.h>
#include <hip/hip_bf16.h>
#include <stdint.h>

#define B_  4
#define S_  2048
#define D_  512
#define H_  8
#define DK_ 64

typedef __bf16 bf16_t;
typedef bf16_t v8bf  __attribute__((ext_vector_type(8)));
typedef bf16_t v16bf __attribute__((ext_vector_type(16)));
typedef float  v8f   __attribute__((ext_vector_type(8)));

__device__ __forceinline__ v8f zero8() {
  v8f z = {0.f,0.f,0.f,0.f,0.f,0.f,0.f,0.f};
  return z;
}

// D = A(16x32 bf16) * B(32x16 bf16) + C(16x16 f32), wave32 WMMA
__device__ __forceinline__ v8f wmma_bf16x2(v8bf a0, v8bf a1, v8bf b0, v8bf b1, v8f c) {
  union { v16bf v; v8bf h[2]; } A, Bm;
  A.h[0] = a0;  A.h[1] = a1;
  Bm.h[0] = b0; Bm.h[1] = b1;
  return __builtin_amdgcn_wmma_f32_16x16x32_bf16(false, A.v, false, Bm.v,
                                                 (short)0, c, false, false);
}

// ---------------------------------------------------------------------------
// Kernel 0: convert + pre-swizzle weights into WMMA B-fragment order (bf16).
// Fragment order: frag(kt, nt) laid out as [lane(0..31)][e(0..15)], where
//   n = nt*16 + (lane&15)
//   k = kt*32 + ((e>>1)/4)*16 + (lane>>4)*8 + 2*((e>>1)&3) + (e&1)
// ---------------------------------------------------------------------------
__global__ __launch_bounds__(256) void pack_weights_kernel(
    const float* __restrict__ Wq, const float* __restrict__ Wk,
    const float* __restrict__ Wv, const float* __restrict__ fc,
    bf16_t* __restrict__ WqP, bf16_t* __restrict__ WkP,
    bf16_t* __restrict__ WvP, bf16_t* __restrict__ fcP) {
  int idx = blockIdx.x * 256 + threadIdx.x;        // 0 .. 4*262144
  int which = idx >> 18;
  int o = idx & 262143;
  int e = o & 15, lane = (o >> 4) & 31, rest = o >> 9;
  int v = e >> 1, pair = e & 1, hf = lane >> 4;
  int kin = ((v >> 2) << 4) + hf * 8 + ((v & 3) << 1) + pair;  // k within 32
  if (which < 3) {
    // W[h,512,64] -> WP[((h*16+kt)*4+nt)*512 + lane*16 + e]
    const float* W = (which == 0) ? Wq : ((which == 1) ? Wk : Wv);
    bf16_t* P = (which == 0) ? WqP : ((which == 1) ? WkP : WvP);
    int nt = rest & 3, kt = (rest >> 2) & 15, h = rest >> 6;
    int n = nt * 16 + (lane & 15);
    int k = kt * 32 + kin;
    P[o] = (bf16_t)W[(h * 512 + k) * 64 + n];
  } else {
    // fc[512,512] (out,in); B(k,n)=fc[n*512+k] -> fcP[(kt*32+nt)*512 + lane*16 + e]
    int nt = rest & 31, kt = rest >> 5;
    int n = nt * 16 + (lane & 15);
    int k = kt * 32 + kin;
    fcP[o] = (bf16_t)fc[n * 512 + k];
  }
}

// ---------------------------------------------------------------------------
// Kernel 1: per-head Q/K/V projections.
// Block = 32 rows of one batch; LDS-stage inputs (bf16, stride 520).
// Outputs: Qb,Kb [B,H,S,64] bf16 ; Vt [B,H,64,S] bf16 (transposed).
// ---------------------------------------------------------------------------
__global__ __launch_bounds__(256) void proj_kernel(
    const float* __restrict__ inQ, const float* __restrict__ inK,
    const float* __restrict__ inV,
    const bf16_t* __restrict__ WqP, const bf16_t* __restrict__ WkP,
    const bf16_t* __restrict__ WvP,
    bf16_t* __restrict__ Qb, bf16_t* __restrict__ Kb, bf16_t* __restrict__ Vt) {
  extern __shared__ char smem[];
  bf16_t* sQ = (bf16_t*)smem;          // 32 x 520
  bf16_t* sK = sQ + 32 * 520;
  bf16_t* sV = sK + 32 * 520;

  int tid = threadIdx.x, lane = tid & 31, wv = tid >> 5;
  int lane15 = lane & 15, hf = lane >> 4;
  int b = blockIdx.x >> 6;             // S/32 = 64 tiles per batch
  int s0 = (blockIdx.x & 63) * 32;

  for (int i = tid; i < 32 * 512; i += 256) {
    int r = i >> 9, c = i & 511;
    size_t g = ((size_t)b * S_ + s0 + r) * 512 + c;
    sQ[r * 520 + c] = (bf16_t)inQ[g];
    sK[r * 520 + c] = (bf16_t)inK[g];
    sV[r * 520 + c] = (bf16_t)inV[g];
  }
  __syncthreads();

  for (int t = wv; t < 192; t += 8) {            // 3 mats * 8 heads * 2 Mt * 4 Nt
    int mat = t >> 6, r = t & 63;
    int h = r >> 3, mt = (r >> 2) & 1, nt = r & 3;
    const bf16_t* sIn = (mat == 0) ? sQ : ((mat == 1) ? sK : sV);
    const bf16_t* WP  = (mat == 0) ? WqP : ((mat == 1) ? WkP : WvP);
    int m = mt * 16 + lane15;
    v8f acc = zero8();
    for (int kt = 0; kt < 16; ++kt) {
      const bf16_t* ar = sIn + m * 520 + kt * 32 + hf * 8;
      v8bf a0 = *(const v8bf*)ar;
      v8bf a1 = *(const v8bf*)(ar + 16);
      const bf16_t* bp = WP + (((h * 16 + kt) * 4 + nt) << 9) + lane * 16;
      v8bf b0 = *(const v8bf*)bp;
      v8bf b1 = *(const v8bf*)(bp + 8);
      acc = wmma_bf16x2(a0, a1, b0, b1, acc);
    }
    int n = nt * 16 + lane15;
    for (int rr = 0; rr < 8; ++rr) {
      int rowt = mt * 16 + rr + 8 * hf;
      bf16_t val = (bf16_t)acc[rr];
      if (mat == 2)
        Vt[((size_t)(b * H_ + h) * DK_ + n) * S_ + (s0 + rowt)] = val;
      else {
        bf16_t* dst = (mat == 0) ? Qb : Kb;
        dst[((size_t)(b * H_ + h) * S_ + s0 + rowt) * DK_ + n] = val;
      }
    }
  }
}

// ---------------------------------------------------------------------------
// Kernel 2: attention. Block = (16 queries, one head, one batch).
// scores tile in LDS (16 x 2052 f32), mask+softmax, write attn once,
// reuse probs for attn @ V (split-K across wave pairs, ds_add_f32 reduce).
// ---------------------------------------------------------------------------
__global__ __launch_bounds__(256) void attn_kernel(
    const bf16_t* __restrict__ Qb, const bf16_t* __restrict__ Kb,
    const bf16_t* __restrict__ Vt, const unsigned char* __restrict__ mask,
    float* __restrict__ attnO, bf16_t* __restrict__ Cx) {
  extern __shared__ char smem[];
  float* sc   = (float*)smem;          // 16 * 2052
  float* ctxs = sc + 16 * 2052;        // 16 * 64
  float* invs = ctxs + 1024;           // 16

  int tid = threadIdx.x, lane = tid & 31, wv = tid >> 5;
  int lane15 = lane & 15, hf = lane >> 4;
  int q0 = blockIdx.x * 16, h = blockIdx.y, b = blockIdx.z;
  int bh = b * H_ + h;
  const float scale = 0.125f;          // 1/sqrt(64)

  // Q A-fragments (held in registers)
  const bf16_t* qbase = Qb + ((size_t)bh * S_ + q0 + lane15) * DK_ + hf * 8;
  v8bf aq0 = *(const v8bf*)(qbase);
  v8bf aq1 = *(const v8bf*)(qbase + 16);
  v8bf aq2 = *(const v8bf*)(qbase + 32);
  v8bf aq3 = *(const v8bf*)(qbase + 48);

  // Phase A: scores = (Q K^T) * scale  -> LDS
  for (int i = 0; i < 16; ++i) {
    int n0 = (wv * 16 + i) * 16;
    const bf16_t* kb = Kb + ((size_t)bh * S_ + n0 + lane15) * DK_ + hf * 8;
    v8f acc = zero8();
    acc = wmma_bf16x2(aq0, aq1, *(const v8bf*)kb, *(const v8bf*)(kb + 16), acc);
    acc = wmma_bf16x2(aq2, aq3, *(const v8bf*)(kb + 32), *(const v8bf*)(kb + 48), acc);
    for (int rr = 0; rr < 8; ++rr)
      sc[(rr + 8 * hf) * 2052 + n0 + lane15] = acc[rr] * scale;
  }
  for (int i = tid; i < 1024; i += 256) ctxs[i] = 0.f;   // zero context accum
  __syncthreads();

  // Phase B1: apply mask (coalesced uchar4 / float4)
  const unsigned char* mrow0 = mask + ((size_t)b * S_ + q0) * S_;
  for (int i = tid; i < (16 * 2048) / 4; i += 256) {
    int i4 = i << 2, m = i4 >> 11, col = i4 & 2047;
    uchar4 mk = *(const uchar4*)(mrow0 + (size_t)m * S_ + col);
    float* p = sc + m * 2052 + col;
    float4 v = *(float4*)p;
    if (mk.x) v.x = -1e9f;
    if (mk.y) v.y = -1e9f;
    if (mk.z) v.z = -1e9f;
    if (mk.w) v.w = -1e9f;
    *(float4*)p = v;
  }
  __syncthreads();

  // Phase B2: row softmax (16 threads per row, shfl-xor width-16 reductions)
  {
    int row = tid >> 4, seg = tid & 15;
    float* rp = sc + row * 2052 + seg * 128;
    float mx = -3.0e38f;
    for (int c = 0; c < 128; ++c) mx = fmaxf(mx, rp[c]);
    for (int mm = 1; mm < 16; mm <<= 1) mx = fmaxf(mx, __shfl_xor(mx, mm, 16));
    float s = 0.f;
    for (int c = 0; c < 128; ++c) { float e = __expf(rp[c] - mx); rp[c] = e; s += e; }
    for (int mm = 1; mm < 16; mm <<= 1) s += __shfl_xor(s, mm, 16);
    if (seg == 0) invs[row] = 1.0f / s;
  }
  __syncthreads();

  // Phase B3: normalize in LDS and stream attn to HBM (float4)
  float* aOut = attnO + ((size_t)bh * S_ + q0) * S_;
  for (int i = tid; i < (16 * 2048) / 4; i += 256) {
    int i4 = i << 2, m = i4 >> 11, col = i4 & 2047;
    float inv = invs[m];
    float* p = sc + m * 2052 + col;
    float4 v = *(float4*)p;
    v.x *= inv; v.y *= inv; v.z *= inv; v.w *= inv;
    *(float4*)p = v;
    *(float4*)(aOut + (size_t)m * S_ + col) = v;
  }
  __syncthreads();

  // Phase C: context = attn @ V ; waves (kh,nt): kh = wv>>2 split-K halves
  {
    int nt = wv & 3, kh = wv >> 2;
    v8f acc = zero8();
    const bf16_t* vtb = Vt + ((size_t)bh * DK_ + nt * 16 + lane15) * S_;
    for (int kk = 0; kk < 32; ++kk) {
      int keyb = (kh * 32 + kk) * 32;
      const float* ap = sc + lane15 * 2052 + keyb + hf * 8;
      v8bf a0, a1;
      for (int e = 0; e < 8; ++e) { a0[e] = (bf16_t)ap[e]; a1[e] = (bf16_t)ap[16 + e]; }
      const bf16_t* vp = vtb + keyb + hf * 8;
      acc = wmma_bf16x2(a0, a1, *(const v8bf*)vp, *(const v8bf*)(vp + 16), acc);
    }
    for (int rr = 0; rr < 8; ++rr)
      atomicAdd(&ctxs[(rr + 8 * hf) * 64 + nt * 16 + lane15], acc[rr]);
  }
  __syncthreads();

  // write context (already [q, h*64+d] transposed layout)
  for (int i = tid; i < 1024; i += 256) {
    int m = i >> 6, d = i & 63;
    Cx[((size_t)b * S_ + q0 + m) * D_ + h * DK_ + d] = (bf16_t)ctxs[i];
  }
}

// ---------------------------------------------------------------------------
// Kernel 3: y = LayerNorm(context @ fc_w^T + input_Q). Block = 16 rows.
// ---------------------------------------------------------------------------
__global__ __launch_bounds__(256) void fc_ln_kernel(
    const bf16_t* __restrict__ Cx, const bf16_t* __restrict__ fcP,
    const float* __restrict__ inQ, float* __restrict__ yOut) {
  extern __shared__ char smem[];
  float* xs = (float*)smem;            // 16 x 520
  int tid = threadIdx.x, lane = tid & 31, wv = tid >> 5;
  int lane15 = lane & 15, hf = lane >> 4;
  int row0 = blockIdx.x * 16;

  v8f acc[4];
  for (int j = 0; j < 4; ++j) acc[j] = zero8();
  const bf16_t* arow = Cx + (size_t)(row0 + lane15) * 512;
  for (int kt = 0; kt < 16; ++kt) {
    const bf16_t* ar = arow + kt * 32 + hf * 8;
    v8bf a0 = *(const v8bf*)ar;
    v8bf a1 = *(const v8bf*)(ar + 16);
    for (int j = 0; j < 4; ++j) {
      int nt = wv * 4 + j;
      const bf16_t* bp = fcP + ((size_t)(kt * 32 + nt) << 9) + lane * 16;
      acc[j] = wmma_bf16x2(a0, a1, *(const v8bf*)bp, *(const v8bf*)(bp + 8), acc[j]);
    }
  }
  for (int j = 0; j < 4; ++j) {
    int n = (wv * 4 + j) * 16 + lane15;
    for (int rr = 0; rr < 8; ++rr)
      xs[(rr + 8 * hf) * 520 + n] = acc[j][rr];
  }
  __syncthreads();

  // residual + LayerNorm: 16 threads per row, 32 cols each
  int row = tid >> 4, seg = tid & 15;
  float* xr = xs + row * 520;
  const float* q = inQ + (size_t)(row0 + row) * 512;
  float s = 0.f;
  for (int c = seg * 32; c < seg * 32 + 32; ++c) {
    float v = xr[c] + q[c];
    xr[c] = v;
    s += v;
  }
  for (int mm = 1; mm < 16; mm <<= 1) s += __shfl_xor(s, mm, 16);
  float mean = s * (1.0f / 512.0f);
  float vs = 0.f;
  for (int c = seg * 32; c < seg * 32 + 32; ++c) {
    float d = xr[c] - mean;
    vs += d * d;
  }
  for (int mm = 1; mm < 16; mm <<= 1) vs += __shfl_xor(vs, mm, 16);
  float rstd = rsqrtf(vs * (1.0f / 512.0f) + 1e-5f);
  float* yo = yOut + (size_t)(row0 + row) * 512;
  for (int c = seg * 32; c < seg * 32 + 32; ++c)
    yo[c] = (xr[c] - mean) * rstd;
}

// ---------------------------------------------------------------------------
extern "C" void kernel_launch(void* const* d_in, const int* in_sizes, int n_in,
                              void* d_out, int out_size, void* d_ws, size_t ws_size,
                              hipStream_t stream) {
  (void)in_sizes; (void)n_in; (void)out_size; (void)ws_size;
  const float* inQ = (const float*)d_in[0];
  const float* inK = (const float*)d_in[1];
  const float* inV = (const float*)d_in[2];
  const unsigned char* mask = (const unsigned char*)d_in[3];   // jnp.bool_ = 1 byte
  const float* Wq = (const float*)d_in[4];
  const float* Wk = (const float*)d_in[5];
  const float* Wv = (const float*)d_in[6];
  const float* fc = (const float*)d_in[7];

  float* yOut  = (float*)d_out;
  float* attnO = (float*)d_out + (size_t)B_ * S_ * D_;         // 4,194,304 floats of y first

  char* ws = (char*)d_ws;
  bf16_t* Qb  = (bf16_t*)(ws);                   //  8 MB  [B,H,S,64]
  bf16_t* Kb  = (bf16_t*)(ws + 8388608);         //  8 MB  [B,H,S,64]
  bf16_t* Vt  = (bf16_t*)(ws + 16777216);        //  8 MB  [B,H,64,S]
  bf16_t* Cx  = (bf16_t*)(ws + 25165824);        //  8 MB  [B,S,512]
  bf16_t* WqP = (bf16_t*)(ws + 33554432);        // 512 KB packed
  bf16_t* WkP = (bf16_t*)(ws + 34078720);
  bf16_t* WvP = (bf16_t*)(ws + 34603008);
  bf16_t* fcP = (bf16_t*)(ws + 35127296);

  pack_weights_kernel<<<4096, 256, 0, stream>>>(Wq, Wk, Wv, fc, WqP, WkP, WvP, fcP);

  size_t lds1 = 3u * 32u * 520u * sizeof(bf16_t);              // 99,840 B
  proj_kernel<<<B_ * (S_ / 32), 256, lds1, stream>>>(inQ, inK, inV,
                                                     WqP, WkP, WvP, Qb, Kb, Vt);

  size_t lds2 = (16u * 2052u + 1024u + 16u) * sizeof(float);   // 135,488 B
  attn_kernel<<<dim3(S_ / 16, H_, B_), 256, lds2, stream>>>(Qb, Kb, Vt, mask,
                                                            attnO, Cx);

  size_t lds3 = 16u * 520u * sizeof(float);                    // 33,280 B
  fc_ln_kernel<<<(B_ * S_) / 16, 256, lds3, stream>>>(Cx, fcP, inQ, yOut);
}
